// LBEBM_1563368095893
// MI455X (gfx1250) — matrix operand order repaired and checked by reading, not compiled, not run on hardware
//
#include <hip/hip_runtime.h>
#include <hip/hip_fp16.h>
#include <math.h>

// ---------------------------------------------------------------------------
// Types for CDNA5 WMMA / TDM
// ---------------------------------------------------------------------------
typedef __attribute__((ext_vector_type(16))) _Float16 f16x16;
typedef __attribute__((ext_vector_type(8)))  _Float16 f16x8;
typedef __attribute__((ext_vector_type(8)))  float    f32x8;
typedef __attribute__((ext_vector_type(4)))  unsigned int u32x4;
typedef __attribute__((ext_vector_type(8)))  int      i32x8;
typedef __attribute__((ext_vector_type(4)))  int      i32x4;

#define HD   224   // EBM hidden 200 padded to 14*16
#define NROW 16    // rows per workgroup tile

// ---- workspace layout (element offsets, _Float16 units) -------------------
// Every GEMM's B matrix is stored transposed (B^T, [N][K] row-major) so that
// a B fragment = two contiguous 16B loads per lane.
#define OFF_W1B   0u        // [224][32]   fwd a1:   B=w1,   B^T=w1^T
#define OFF_W2T   7168u     // [224][224]  fwd a2:   B=w2,   B^T=w2^T
#define OFF_W2    57344u    // [224][224]  bwd dh1:  B=w2^T, B^T=w2
#define OFF_W3T   107520u   // [32][224]   fwd logit:B=w3,   B^T=w3^T
#define OFF_W3B   114688u   // [224][32]   bwd dh2:  B=w3^T, B^T=w3
#define OFF_W1Z   121856u   // [16][224]   bwd dh0z: B^T=w1[z-rows]
#define WHALFS    125440u   // end of langevin weight pack (250880 B contiguous)
#define OFF_DW1T  125440u   // [1024][32]
#define OFF_DW2T  158208u   // [512][1024]
#define OFF_DW3T  682496u   // [1024][512]
#define OFF_DW4T  1206784u  // [16][1024]
#define OFF_ZK    1223168u  // [N][16] f16 z_k
#define OFF_BIAS  2271744u  // f32 region: padded biases (3056 floats)
// bias sub-offsets (f32 units inside bias region)
#define BB1   0      // [224]
#define BB2   224    // [224]
#define BB3   448    // [32]
#define BD1   480    // [1024]
#define BD2   1504   // [512]
#define BD3   2016   // [1024]
#define BD4   3040   // [16]
#define NB_EBM 480
#define NB_DEC 2576

// ---------------------------------------------------------------------------
// Fragment loaders (layouts per CDNA5 ISA 7.12.2)
// A 16x32 f16: lane m=lane&15, h=lane>>4; K = {k0+8h..+8} and {k0+16+8h..+8}
// B^T [N][K]:  lane n=lane&15, h=lane>>4; K = {k0+8h..+8} and {k0+16+8h..+8}
// ---------------------------------------------------------------------------
__device__ inline f16x16 ldA(const _Float16* p, int ld, int k0) {
  int lane = threadIdx.x & 31;
  int m = lane & 15, h = lane >> 4;
  const _Float16* q = p + m * ld + k0 + 8 * h;
  f16x8 lo = *(const f16x8*)q;
  f16x8 hi = *(const f16x8*)(q + 16);
  f16x16 a;
#pragma unroll
  for (int i = 0; i < 8; ++i) { a[i] = lo[i]; a[i + 8] = hi[i]; }
  return a;
}

__device__ inline f16x16 ldB(const _Float16* bt, int ld, int n0, int k0) {
  int lane = threadIdx.x & 31;
  int n = lane & 15, h = lane >> 4;
  const _Float16* q = bt + (size_t)(n0 + n) * ld + k0 + 8 * h;
  f16x8 lo = *(const f16x8*)q;
  f16x8 hi = *(const f16x8*)(q + 16);
  f16x16 b;
#pragma unroll
  for (int i = 0; i < 8; ++i) { b[i] = lo[i]; b[i + 8] = hi[i]; }
  return b;
}

__device__ inline f32x8 wmma16(f16x16 a, f16x16 b, f32x8 c) {
  return __builtin_amdgcn_wmma_f32_16x16x32_f16(false, a, false, b,
                                                (short)0, c, false, false);
}

// ---------------------------------------------------------------------------
// Branch-free erf (Abramowitz & Stegun 7.1.26): one fast rcp + one fast exp,
// pure FMA chain, no exec-mask divergence.
// ---------------------------------------------------------------------------
__device__ inline float erf_bf(float x) {
  float ax = fabsf(x);
  float t = __builtin_amdgcn_rcpf(__builtin_fmaf(0.3275911f, ax, 1.0f));
  float y = 1.061405429f;
  y = __builtin_fmaf(y, t, -1.453152027f);
  y = __builtin_fmaf(y, t, 1.421413741f);
  y = __builtin_fmaf(y, t, -0.284496736f);
  y = __builtin_fmaf(y, t, 0.254829592f);
  y = y * t;
  float e = __expf(-ax * ax);
  float r = __builtin_fmaf(-y, e, 1.0f);
  return copysignf(r, x);
}
__device__ inline float gelu_f(float x) {
  return 0.5f * x * (1.0f + erf_bf(x * 0.70710678118654752f));
}
__device__ inline float gelu_g(float x) {
  float cdf = 0.5f * (1.0f + erf_bf(x * 0.70710678118654752f));
  float pdf = 0.39894228040143268f * __expf(-0.5f * x * x);
  return __builtin_fmaf(x, pdf, cdf);
}

// ---------------------------------------------------------------------------
// TDM: stage the 250880B langevin weight pack (global -> LDS offset 0) as a
// 2D tile of 8-byte elements: tile_dim0 = 3136, tile_dim1 = 10.
// D# built per CDNA5 ISA ch.8 (group0: count/lds/global/type, group1: dims).
// This toolchain exposes the 6-arg builtin:
//   (u32x4 g0, i32x8 g1, i32x4 g2, i32x4 g3, i32x8 g4, i32 cpol)
// ---------------------------------------------------------------------------
#if __has_builtin(__builtin_amdgcn_tensor_load_to_lds)
#define HAVE_TDM 1
__device__ inline void tdm_stage_weights(const _Float16* wpack) {
  unsigned long long ga = (unsigned long long)(uintptr_t)wpack;
  u32x4 g0;
  g0[0] = 1u;                                        // count=1, user D#
  g0[1] = 0u;                                        // lds_addr = 0 (dyn LDS base)
  g0[2] = (unsigned)(ga & 0xFFFFFFFFull);            // global_addr[31:0]
  g0[3] = (unsigned)((ga >> 32) & 0x01FFFFFFull)     // global_addr[56:32]
          | (2u << 30);                              // type = 2 ("image")
  i32x8 g1;
  g1[0] = (int)(3u << 16);        // data_size = 3 (8B), wg_mask=0, no flags
  g1[1] = (int)(3136u << 16);     // tensor_dim0[15:0] (bits 63:48)
  g1[2] = (int)(10u << 16);       // tensor_dim1[15:0]
  g1[3] = (int)(3136u << 16);     // tile_dim0 (bits 127:112)
  g1[4] = 10;                     // tile_dim1 (bits 143:128), tile_dim2=0
  g1[5] = 3136;                   // tensor_dim0_stride[31:0]
  g1[6] = 0;                      // stride hi / dim1_stride lo
  g1[7] = 0;
  i32x4 gz4 = {0, 0, 0, 0};
  i32x8 gz8 = {0, 0, 0, 0, 0, 0, 0, 0};
  __builtin_amdgcn_tensor_load_to_lds(g0, g1, gz4, gz4, gz8, 0);
}
#else
#define HAVE_TDM 0
#endif

// ---------------------------------------------------------------------------
// Prep: pad / (optionally) transpose an f32 weight into an f16 B^T pack.
// ---------------------------------------------------------------------------
__global__ void packW(const float* __restrict__ src, int ld, int sr, int sc,
                      _Float16* __restrict__ dst, int dr, int dc, int trans) {
  int i = blockIdx.x * blockDim.x + threadIdx.x;
  if (i >= dr * dc) return;
  int r = i / dc, c = i % dc;
  float v = 0.0f;
  if (!trans) { if (r < sr && c < sc) v = src[(size_t)r * ld + c]; }
  else        { if (c < sr && r < sc) v = src[(size_t)c * ld + r]; }
  dst[i] = (_Float16)v;
}

// Pad an f32 bias vector into the f32 bias region (zeros beyond n).
__global__ void packB(const float* __restrict__ src, int n,
                      float* __restrict__ dst, int np) {
  int i = blockIdx.x * blockDim.x + threadIdx.x;
  if (i < np) dst[i] = (i < n) ? src[i] : 0.0f;
}

// ---------------------------------------------------------------------------
// Langevin: 20 steps of grad-based sampling. One 16-row tile per workgroup,
// 8 waves split the 14 N-tiles of the 224-wide hidden dim. EBM weights
// (TDM-staged), padded biases and all activations persist in LDS.
// ---------------------------------------------------------------------------
__global__ __launch_bounds__(256, 1)
void langevin_kernel(const float* __restrict__ ftraj, const float* __restrict__ z0,
                     const float* __restrict__ noise,
                     const float* __restrict__ biaspack,   // padded EBM biases
                     const _Float16* __restrict__ wpack,
                     _Float16* __restrict__ zk, float* __restrict__ outF,
                     int nrows) {
  extern __shared__ char smem_raw[];
  _Float16* sw = (_Float16*)smem_raw;

  const int t = threadIdx.x;
  const int wv = t >> 5, lane = t & 31, h = lane >> 4, col = lane & 15;
  const int base = blockIdx.x * NROW;

#if HAVE_TDM
  // One DMA descriptor, issued by wave 0; EXEC-independent, TENSORcnt-tracked.
  if (wv == 0) {
    tdm_stage_weights(wpack);
    __builtin_amdgcn_s_wait_tensorcnt(0);
  }
#else
  {
    const int4* src = (const int4*)wpack;
    int4* dst = (int4*)sw;
    for (int i = t; i < (int)(WHALFS / 8); i += 256) dst[i] = src[i];
  }
#endif

  _Float16* sW1B = sw + OFF_W1B;   // [224][32]
  _Float16* sW2T = sw + OFF_W2T;   // [224][224]
  _Float16* sW2  = sw + OFF_W2;    // [224][224]
  _Float16* sW3T = sw + OFF_W3T;   // [32][224]
  _Float16* sW3B = sw + OFF_W3B;   // [224][32]
  _Float16* sW1Z = sw + OFF_W1Z;   // [16][224]

  _Float16* act  = sw + WHALFS;
  _Float16* sH0  = act;            // [16][32]
  _Float16* sA1  = act + 512;      // [16][224] pre-act
  _Float16* sH1  = sA1 + 3584;     // [16][224] gelu
  _Float16* sA2  = sH1 + 3584;
  _Float16* sH2  = sA2 + 3584;
  _Float16* sD2  = sH2 + 3584;     // d a2
  _Float16* sD1  = sD2 + 3584;     // d a1
  _Float16* sDL  = sD1 + 3584;     // [16][32] dlogits f16
  _Float16* sC16 = sDL + 512;      // [16][16] cond f16
  float* sZ  = (float*)(sC16 + 256);  // [16][16]
  float* sLG = sZ + 256;              // [16][32] logits f32
  float* sG  = sLG + 512;             // [16][16] grad wrt z
  float* sB1 = sG + 256;              // [224]
  float* sB2 = sB1 + 224;             // [224]
  float* sB3 = sB2 + 224;             // [32]

  for (int i = t; i < NB_EBM; i += 256) sB1[i] = biaspack[i];
  sC16[t] = (_Float16)ftraj[(size_t)base * 16 + t];
  sZ[t]   = z0[(size_t)base * 16 + t];
  __syncthreads();

  const float step2 = 0.5f * 0.4f * 0.4f;   // 0.08
  const float inv_sig2 = 0.25f;

  for (int step = 0; step < 20; ++step) {
    // h0 = [z | cond] in f16
    for (int e = t; e < 512; e += 256) {
      int r = e >> 5, c = e & 31;
      sH0[e] = (c < 16) ? (_Float16)sZ[r * 16 + c] : sC16[r * 16 + (c - 16)];
    }
    __syncthreads();

    // GEMM1: a1 = h0 @ w1 + b1 ; h1 = gelu(a1)
    {
      f16x16 a0 = ldA(sH0, 32, 0);
      for (int tile = wv; tile < 14; tile += 8) {
        f32x8 c = {};
        c = wmma16(a0, ldB(sW1B, 32, tile * 16, 0), c);
        int n0 = tile * 16;
#pragma unroll
        for (int r = 0; r < 8; ++r) {
          int row = r + 8 * h, j = n0 + col;
          float x = c[r] + sB1[j];
          sA1[row * HD + j] = (_Float16)x;
          sH1[row * HD + j] = (_Float16)gelu_f(x);
        }
      }
    }
    __syncthreads();

    // GEMM2: a2 = h1 @ w2 + b2 ; h2 = gelu(a2)
    for (int tile = wv; tile < 14; tile += 8) {
      f32x8 c = {};
#pragma unroll
      for (int k0 = 0; k0 < HD; k0 += 32)
        c = wmma16(ldA(sH1, HD, k0), ldB(sW2T, HD, tile * 16, k0), c);
      int n0 = tile * 16;
#pragma unroll
      for (int r = 0; r < 8; ++r) {
        int row = r + 8 * h, j = n0 + col;
        float x = c[r] + sB2[j];
        sA2[row * HD + j] = (_Float16)x;
        sH2[row * HD + j] = (_Float16)gelu_f(x);
      }
    }
    __syncthreads();

    // GEMM3: logits = h2 @ w3 + b3
    if (wv < 2) {
      int tile = wv;
      f32x8 c = {};
#pragma unroll
      for (int k0 = 0; k0 < HD; k0 += 32)
        c = wmma16(ldA(sH2, HD, k0), ldB(sW3T, HD, tile * 16, k0), c);
      int n0 = tile * 16;
#pragma unroll
      for (int r = 0; r < 8; ++r) {
        int row = r + 8 * h, j = n0 + col;
        sLG[row * 32 + j] = c[r] + sB3[j];
      }
    }
    __syncthreads();

    // d logits = -softmax(logits) over the 20 valid columns
    if (t < 16) {
      float m = -1e30f;
      for (int j = 0; j < 20; ++j) m = fmaxf(m, sLG[t * 32 + j]);
      float s = 0.0f;
      for (int j = 0; j < 20; ++j) s += __expf(sLG[t * 32 + j] - m);
      float inv = __builtin_amdgcn_rcpf(s);
      for (int j = 0; j < 32; ++j)
        sDL[t * 32 + j] = (j < 20)
            ? (_Float16)(-__expf(sLG[t * 32 + j] - m) * inv)
            : (_Float16)0.0f;
    }
    __syncthreads();

    // dh2 = dlogits @ w3^T ; da2 = dh2 * gelu'(a2)
    {
      f16x16 adl = ldA(sDL, 32, 0);
      for (int tile = wv; tile < 14; tile += 8) {
        f32x8 c = {};
        c = wmma16(adl, ldB(sW3B, 32, tile * 16, 0), c);
        int n0 = tile * 16;
#pragma unroll
        for (int r = 0; r < 8; ++r) {
          int row = r + 8 * h, j = n0 + col;
          sD2[row * HD + j] = (_Float16)(c[r] * gelu_g((float)sA2[row * HD + j]));
        }
      }
    }
    __syncthreads();

    // dh1 = da2 @ w2^T ; da1 = dh1 * gelu'(a1)
    for (int tile = wv; tile < 14; tile += 8) {
      f32x8 c = {};
#pragma unroll
      for (int k0 = 0; k0 < HD; k0 += 32)
        c = wmma16(ldA(sD2, HD, k0), ldB(sW2, HD, tile * 16, k0), c);
      int n0 = tile * 16;
#pragma unroll
      for (int r = 0; r < 8; ++r) {
        int row = r + 8 * h, j = n0 + col;
        sD1[row * HD + j] = (_Float16)(c[r] * gelu_g((float)sA1[row * HD + j]));
      }
    }
    __syncthreads();

    // g = da1 @ w1[z-rows]^T  -> [16,16]
    if (wv == 0) {
      f32x8 c = {};
#pragma unroll
      for (int k0 = 0; k0 < HD; k0 += 32)
        c = wmma16(ldA(sD1, HD, k0), ldB(sW1Z, HD, 0, k0), c);
#pragma unroll
      for (int r = 0; r < 8; ++r) sG[(r + 8 * h) * 16 + col] = c[r];
    }
    __syncthreads();

    // Langevin update
    {
      int r = t >> 4, cc = t & 15;
      float eps = noise[((size_t)step * nrows + base + r) * 16 + cc];
      float z = sZ[t];
      z = z - step2 * (sG[t] + inv_sig2 * z) + 0.4f * eps;
      sZ[t] = z;
    }
    __syncthreads();
  }

  zk[(size_t)base * 16 + t] = (_Float16)sZ[t];
  // ftraj passthrough: second tuple output
  outF[(size_t)nrows * 10 + (size_t)base * 16 + t] = ftraj[(size_t)base * 16 + t];
}

// ---------------------------------------------------------------------------
// Decoder: [ftraj|z_k] -> 1024 relu -> 512 relu -> 1024 relu -> 10.
// Activations + padded biases in LDS, weight B^T fragments streamed from
// global (L2-resident, 2.2MB) with prefetch of the next tile.
// ---------------------------------------------------------------------------
__global__ __launch_bounds__(256, 1)
void decoder_kernel(const float* __restrict__ ftraj, const _Float16* __restrict__ zk,
                    const _Float16* __restrict__ dw1t, const _Float16* __restrict__ dw2t,
                    const _Float16* __restrict__ dw3t, const _Float16* __restrict__ dw4t,
                    const float* __restrict__ biaspack,  // padded dec biases
                    float* __restrict__ out, int nrows) {
  extern __shared__ char smem_raw[];
  _Float16* sIN = (_Float16*)smem_raw;   // [16][32]
  _Float16* sX1 = sIN + 512;             // [16][1024]
  _Float16* sX2 = sX1 + 16384;           // [16][512]
  _Float16* sX3 = sX2 + 8192;            // [16][1024]
  float*    sO  = (float*)(sX3 + 16384); // [16][16]
  float*    sDB1 = sO + 256;             // [1024]
  float*    sDB2 = sDB1 + 1024;          // [512]
  float*    sDB3 = sDB2 + 512;           // [1024]
  float*    sDB4 = sDB3 + 1024;          // [16]

  const int t = threadIdx.x;
  const int wv = t >> 5, lane = t & 31, h = lane >> 4, col = lane & 15;
  const int base = blockIdx.x * NROW;

  for (int i = t; i < NB_DEC; i += 256) sDB1[i] = biaspack[i];
  for (int e = t; e < 512; e += 256) {
    int r = e >> 5, c = e & 31;
    sIN[e] = (c < 16) ? (_Float16)ftraj[(size_t)(base + r) * 16 + c]
                      : zk[(size_t)(base + r) * 16 + (c - 16)];
  }
  __syncthreads();

  // L1: [16,32] @ [32,1024] + b, relu
  {
    f16x16 a0 = ldA(sIN, 32, 0);
    for (int tile = wv; tile < 64; tile += 8) {
      f32x8 c = {};
      c = wmma16(a0, ldB(dw1t, 32, tile * 16, 0), c);
      int n0 = tile * 16;
#pragma unroll
      for (int r = 0; r < 8; ++r) {
        int row = r + 8 * h, j = n0 + col;
        sX1[row * 1024 + j] = (_Float16)fmaxf(c[r] + sDB1[j], 0.0f);
      }
    }
  }
  __syncthreads();

  // L2: [16,1024] @ [1024,512] + b, relu
  for (int tile = wv; tile < 32; tile += 8) {
    if (tile + 8 < 32)  // prefetch next tile's B^T rows (global_prefetch_b8)
      __builtin_prefetch(dw2t + (size_t)((tile + 8) * 16 + col) * 1024, 0, 0);
    f32x8 c = {};
#pragma unroll 8
    for (int k0 = 0; k0 < 1024; k0 += 32)
      c = wmma16(ldA(sX1, 1024, k0), ldB(dw2t, 1024, tile * 16, k0), c);
    int n0 = tile * 16;
#pragma unroll
    for (int r = 0; r < 8; ++r) {
      int row = r + 8 * h, j = n0 + col;
      sX2[row * 512 + j] = (_Float16)fmaxf(c[r] + sDB2[j], 0.0f);
    }
  }
  __syncthreads();

  // L3: [16,512] @ [512,1024] + b, relu
  for (int tile = wv; tile < 64; tile += 8) {
    if (tile + 8 < 64)
      __builtin_prefetch(dw3t + (size_t)((tile + 8) * 16 + col) * 512, 0, 0);
    f32x8 c = {};
#pragma unroll 8
    for (int k0 = 0; k0 < 512; k0 += 32)
      c = wmma16(ldA(sX2, 512, k0), ldB(dw3t, 512, tile * 16, k0), c);
    int n0 = tile * 16;
#pragma unroll
    for (int r = 0; r < 8; ++r) {
      int row = r + 8 * h, j = n0 + col;
      sX3[row * 1024 + j] = (_Float16)fmaxf(c[r] + sDB3[j], 0.0f);
    }
  }
  __syncthreads();

  // L4: [16,1024] @ [1024,16(pad 10)] + b
  if (wv == 0) {
    f32x8 c = {};
#pragma unroll 8
    for (int k0 = 0; k0 < 1024; k0 += 32)
      c = wmma16(ldA(sX3, 1024, k0), ldB(dw4t, 1024, 0, k0), c);
#pragma unroll
    for (int r = 0; r < 8; ++r) {
      int row = r + 8 * h, j = col;
      sO[row * 16 + j] = c[r] + sDB4[j];
    }
  }
  __syncthreads();

  if (t < 160) {
    int r = t / 10, c = t % 10;
    out[(size_t)(base + r) * 10 + c] = sO[r * 16 + c];
  }
}

// ---------------------------------------------------------------------------
extern "C" void kernel_launch(void* const* d_in, const int* in_sizes, int n_in,
                              void* d_out, int out_size, void* d_ws, size_t ws_size,
                              hipStream_t stream) {
  const float* ftraj = (const float*)d_in[0];
  const float* z0    = (const float*)d_in[1];
  const float* noise = (const float*)d_in[2];
  const float* ew1 = (const float*)d_in[3];
  const float* eb1 = (const float*)d_in[4];
  const float* ew2 = (const float*)d_in[5];
  const float* eb2 = (const float*)d_in[6];
  const float* ew3 = (const float*)d_in[7];
  const float* eb3 = (const float*)d_in[8];
  const float* dw1 = (const float*)d_in[9];
  const float* db1 = (const float*)d_in[10];
  const float* dw2 = (const float*)d_in[11];
  const float* db2 = (const float*)d_in[12];
  const float* dw3 = (const float*)d_in[13];
  const float* db3 = (const float*)d_in[14];
  const float* dw4 = (const float*)d_in[15];
  const float* db4 = (const float*)d_in[16];
  float* out = (float*)d_out;
  _Float16* ws = (_Float16*)d_ws;
  float* wsb = (float*)(ws + OFF_BIAS);

  const int nrows = in_sizes[0] / 16;          // 65536

  auto pack = [&](const float* s, int ld, int sr, int sc,
                  size_t off, int dr, int dc, int tr) {
    int tot = dr * dc;
    packW<<<(tot + 255) / 256, 256, 0, stream>>>(s, ld, sr, sc, ws + off, dr, dc, tr);
  };
  // EBM packs (all B^T, [N][K] row-major, zero padded)
  pack(ew1, 200, 32, 200, OFF_W1B, 224, 32, 1);
  pack(ew2, 200, 200, 200, OFF_W2T, 224, 224, 1);
  pack(ew2, 200, 200, 200, OFF_W2,  224, 224, 0);
  pack(ew3, 20, 200, 20, OFF_W3T, 32, 224, 1);
  pack(ew3, 20, 200, 20, OFF_W3B, 224, 32, 0);
  pack(ew1, 200, 16, 200, OFF_W1Z, 16, 224, 0);
  // Decoder packs
  pack(dw1, 1024, 32, 1024, OFF_DW1T, 1024, 32, 1);
  pack(dw2, 512, 1024, 512, OFF_DW2T, 512, 1024, 1);
  pack(dw3, 1024, 512, 1024, OFF_DW3T, 1024, 512, 1);
  pack(dw4, 10, 1024, 10, OFF_DW4T, 16, 1024, 1);
  // Bias packs (padded, zeros beyond valid range)
  packB<<<1, 256, 0, stream>>>(eb1, 200, wsb + BB1, 224);
  packB<<<1, 256, 0, stream>>>(eb2, 200, wsb + BB2, 224);
  packB<<<1, 32, 0, stream>>>(eb3, 20, wsb + BB3, 32);
  packB<<<4, 256, 0, stream>>>(db1, 1024, wsb + BD1, 1024);
  packB<<<2, 256, 0, stream>>>(db2, 512, wsb + BD2, 512);
  packB<<<4, 256, 0, stream>>>(db3, 1024, wsb + BD3, 1024);
  packB<<<1, 16, 0, stream>>>(db4, 10, wsb + BD4, 16);

  const size_t lds_langevin = 302464;  // weights + acts + biases, < 320KB/WGP
  langevin_kernel<<<nrows / NROW, 256, lds_langevin, stream>>>(
      ftraj, z0, noise, wsb, ws, ws + OFF_ZK, out, nrows);

  const size_t lds_decoder = 94272;
  decoder_kernel<<<nrows / NROW, 256, lds_decoder, stream>>>(
      ftraj, ws + OFF_ZK, ws + OFF_DW1T, ws + OFF_DW2T, ws + OFF_DW3T,
      ws + OFF_DW4T, wsb + BD1, out, nrows);

  (void)n_in; (void)out_size; (void)ws_size;
}
